// RSSM_6777458393099
// MI455X (gfx1250) — matrix operand order repaired and checked by reading, not compiled, not run on hardware
//
#include <hip/hip_runtime.h>
#include <math.h>

typedef __attribute__((ext_vector_type(16))) __bf16 v16bf;
typedef __attribute__((ext_vector_type(8)))  __bf16 v8bf;
typedef __attribute__((ext_vector_type(8)))  float  v8f;

#define AS1 __attribute__((address_space(1)))
// global-address-space casts: force global_load/global_store instead of flat_*
#define GCF(p) ((const AS1 float*)(unsigned long long)(p))
#define GCB(p) ((const AS1 __bf16*)(unsigned long long)(p))
#define GF(p)  ((AS1 float*)(unsigned long long)(p))
#define GB(p)  ((AS1 __bf16*)(unsigned long long)(p))

#define CT 128      // T
#define CB 128      // B
#define CEMB 1024
#define CACT 12
#define CSTOCH 32
#define CHID 256
#define CFEAT 512
#define LOG2PI 1.8378770664093453f

// Re-materialize a uniform pointer each loop iteration so LLVM cannot hoist
// the (loop-invariant) weight-fragment loads out of the t-loop and spill them.
#define ANTI_HOIST(p) asm volatile("" : "+s"(p))

// ---------------- WMMA helpers (CDNA5 bf16 16x16x32, f32 accum) ----------------
__device__ inline v8f z8() { v8f z = {0.f,0.f,0.f,0.f,0.f,0.f,0.f,0.f}; return z; }

__device__ inline v8f wmma_bf16(v16bf a, v16bf b, v8f c) {
  return __builtin_amdgcn_wmma_f32_16x16x32_bf16(false, a, false, b, (short)0, c, false, false);
}

// A fragment from LDS: 16x32 bf16 tile, row-major, leading dim lda.
// lane L: m = L&15 ; kb = k0 + (L>>4)*8 ; elems 0-7 -> K=kb.., elems 8-15 -> K=kb+16..
__device__ inline v16bf load_a(const __bf16* A, int lda, int k0, int lane) {
  int m  = lane & 15;
  int kb = k0 + ((lane >> 4) << 3);
  const __bf16* p = A + m * lda + kb;
  v8bf lo = *(const v8bf*)(p);
  v8bf hi = *(const v8bf*)(p + 16);
  v16bf a;
#pragma unroll
  for (int i = 0; i < 8; ++i) { a[i] = lo[i]; a[i + 8] = hi[i]; }
  return a;
}

// A fragment from GLOBAL bf16 (decoder activations)
__device__ inline v16bf load_a_g(const __bf16* A, int lda, int k0, int lane) {
  int m  = lane & 15;
  int kb = k0 + ((lane >> 4) << 3);
  const __bf16* p = A + m * lda + kb;
  v8bf lo = *(const AS1 v8bf*)(unsigned long long)(p);
  v8bf hi = *(const AS1 v8bf*)(unsigned long long)(p + 16);
  v16bf a;
#pragma unroll
  for (int i = 0; i < 8; ++i) { a[i] = lo[i]; a[i + 8] = hi[i]; }
  return a;
}

// A fragment from GLOBAL fp32 (converted on the fly)
__device__ inline v16bf load_a_f32(const float* A, int lda, int k0, int lane) {
  int m  = lane & 15;
  int kb = k0 + ((lane >> 4) << 3);
  const AS1 float* p = GCF(A + m * lda + kb);
  v16bf a;
#pragma unroll
  for (int i = 0; i < 8; ++i) { a[i] = (__bf16)p[i]; a[i + 8] = (__bf16)p[16 + i]; }
  return a;
}

// B fragment: weights stored transposed (N x K row-major) in GLOBAL memory.
// lane L: n = n0 + (L&15) ; kb = k0 + (L>>4)*8 ; elems 0-7 -> K=kb.., elems 8-15 -> K=kb+16..
__device__ inline v16bf load_b(const __bf16* WT, int K, int n0, int k0, int lane) {
  int n  = n0 + (lane & 15);
  int kb = k0 + ((lane >> 4) << 3);
  const __bf16* p = WT + (size_t)n * K + kb;
  v8bf lo = *(const AS1 v8bf*)(unsigned long long)(p);
  v8bf hi = *(const AS1 v8bf*)(unsigned long long)(p + 16);
  v16bf b;
#pragma unroll
  for (int i = 0; i < 8; ++i) { b[i] = lo[i]; b[i + 8] = hi[i]; }
  return b;
}

__device__ inline float sigmf(float x) { return 1.f / (1.f + expf(-x)); }
__device__ inline float softplusf(float x) { return (x > 20.f) ? x : logf(1.f + expf(x)); }
__device__ inline float eluf(float x) { return (x > 0.f) ? x : (expf(x) - 1.f); }

// ---------------- weight transpose + fp32->bf16 (WT[n*Kpad+k] = W[k*N+n]) -----
__global__ void k_transpose(const float* __restrict__ W, __bf16* __restrict__ WT,
                            int K, int N, int Kpad) {
  int idx = blockIdx.x * blockDim.x + threadIdx.x;
  if (idx >= N * Kpad) return;
  int n = idx / Kpad, k = idx - n * Kpad;
  float v = (k < K) ? GCF(W)[(size_t)k * N + n] : 0.f;
  GB(WT)[idx] = (__bf16)v;
}

// ---------------- sequential scan: 8 blocks x 16 batch rows, 16 waves --------
__global__ __launch_bounds__(512) void k_scan(
    const float* __restrict__ emb, const float* __restrict__ action,
    const float* __restrict__ noise,
    const __bf16* __restrict__ wihT, const float* __restrict__ bih,
    const __bf16* __restrict__ whhT, const float* __restrict__ bhh,
    // posterior MLP (input [h,e] K=1280)
    const __bf16* __restrict__ qw0T, const float* __restrict__ qb0,
    const __bf16* __restrict__ qw1T, const float* __restrict__ qb1,
    const __bf16* __restrict__ qwoT, const float* __restrict__ qbo,
    // prior MLP (input h K=256)
    const __bf16* __restrict__ pw0T, const float* __restrict__ pb0,
    const __bf16* __restrict__ pw1T, const float* __restrict__ pb1,
    const __bf16* __restrict__ pwoT, const float* __restrict__ pbo,
    __bf16* __restrict__ states, float* __restrict__ kl_part) {
  __shared__ __bf16 sHbf[16 * 256];   // h (bf16)
  __shared__ __bf16 sXA[16 * 64];     // [s(32), a(12), pad] bf16
  __shared__ __bf16 sHid0[16 * 512];
  __shared__ __bf16 sHid1[16 * 512];
  __shared__ float  sOut[16 * 64];
  __shared__ float  sPM[512];
  __shared__ float  sPS[512];
  __shared__ float  sRed[16];

  const int tid = threadIdx.x, wave = tid >> 5, lane = tid & 31;
  const int row0 = blockIdx.x * 16;
  const int n15 = lane & 15, mb = (lane >> 4) * 8;

  for (int i = tid; i < 16 * 256; i += 512) sHbf[i] = (__bf16)0.f;
  for (int i = tid; i < 16 * 64;  i += 512) sXA[i]  = (__bf16)0.f;
  float klacc = 0.f;
  __syncthreads();

  for (int t = 0; t < CT; ++t) {
    // defeat LICM of the per-step weight-fragment loads (keep them in L2, not VGPRs)
    const __bf16* wih_t = wihT;  ANTI_HOIST(wih_t);
    const __bf16* whh_t = whhT;  ANTI_HOIST(whh_t);
    const __bf16* qw0_t = qw0T;  ANTI_HOIST(qw0_t);
    const __bf16* qw1_t = qw1T;  ANTI_HOIST(qw1_t);
    const __bf16* qwo_t = qwoT;  ANTI_HOIST(qwo_t);
    const __bf16* pw0_t = pw0T;  ANTI_HOIST(pw0_t);
    const __bf16* pw1_t = pw1T;  ANTI_HOIST(pw1_t);
    const __bf16* pwo_t = pwoT;  ANTI_HOIST(pwo_t);

    // stage action for this step
    for (int i = tid; i < 16 * CACT; i += 512) {
      int m = i / CACT, c = i - m * CACT;
      sXA[m * 64 + 32 + c] = (__bf16)GCF(action)[((size_t)t * CB + row0 + m) * CACT + c];
    }
    __syncthreads();

    // ---- GRU: wave owns gate columns [wave*16, wave*16+16) of 256 ----
    const int ncol = wave * 16;
    v8f gir = z8(), giz = z8(), gin = z8();
    v8f ghr = z8(), ghz = z8(), ghn = z8();
#pragma unroll
    for (int k0 = 0; k0 < 64; k0 += 32) {
      v16bf a = load_a(sXA, 64, k0, lane);
      gir = wmma_bf16(a, load_b(wih_t, 64, ncol,       k0, lane), gir);
      giz = wmma_bf16(a, load_b(wih_t, 64, 256 + ncol, k0, lane), giz);
      gin = wmma_bf16(a, load_b(wih_t, 64, 512 + ncol, k0, lane), gin);
    }
    for (int k0 = 0; k0 < 256; k0 += 32) {
      v16bf a = load_a(sHbf, 256, k0, lane);
      ghr = wmma_bf16(a, load_b(whh_t, 256, ncol,       k0, lane), ghr);
      ghz = wmma_bf16(a, load_b(whh_t, 256, 256 + ncol, k0, lane), ghz);
      ghn = wmma_bf16(a, load_b(whh_t, 256, 512 + ncol, k0, lane), ghn);
    }
    float hnew[8];
    {
      int col = ncol + n15;
#pragma unroll
      for (int j = 0; j < 8; ++j) {
        int m = mb + j;
        float ir = gir[j] + GCF(bih)[col];
        float iz = giz[j] + GCF(bih)[256 + col];
        float in_= gin[j] + GCF(bih)[512 + col];
        float hr = ghr[j] + GCF(bhh)[col];
        float hz = ghz[j] + GCF(bhh)[256 + col];
        float hn = ghn[j] + GCF(bhh)[512 + col];
        float r = sigmf(ir + hr), z = sigmf(iz + hz);
        float n = tanhf(in_ + r * hn);
        float hold = (float)sHbf[m * 256 + col];
        hnew[j] = (1.f - z) * n + z * hold;
      }
    }
    __syncthreads();  // all reads of old h done
    {
      int col = ncol + n15;
#pragma unroll
      for (int j = 0; j < 8; ++j) sHbf[(mb + j) * 256 + col] = (__bf16)hnew[j];
    }
    __syncthreads();

    // ---- prior MLP: h -> 512 -> 512 -> 64 ----
    {
      v8f acc0 = z8(), acc1 = z8();
      for (int k0 = 0; k0 < 256; k0 += 32) {
        v16bf a = load_a(sHbf, 256, k0, lane);
        acc0 = wmma_bf16(a, load_b(pw0_t, 256, (wave * 2 + 0) * 16, k0, lane), acc0);
        acc1 = wmma_bf16(a, load_b(pw0_t, 256, (wave * 2 + 1) * 16, k0, lane), acc1);
      }
      int n0 = wave * 32 + n15;
      float b0 = GCF(pb0)[n0], b1 = GCF(pb0)[n0 + 16];
#pragma unroll
      for (int j = 0; j < 8; ++j) {
        sHid0[(mb + j) * 512 + n0]      = (__bf16)eluf(acc0[j] + b0);
        sHid0[(mb + j) * 512 + n0 + 16] = (__bf16)eluf(acc1[j] + b1);
      }
    }
    __syncthreads();
    {
      v8f acc0 = z8(), acc1 = z8();
      for (int k0 = 0; k0 < 512; k0 += 32) {
        v16bf a = load_a(sHid0, 512, k0, lane);
        acc0 = wmma_bf16(a, load_b(pw1_t, 512, (wave * 2 + 0) * 16, k0, lane), acc0);
        acc1 = wmma_bf16(a, load_b(pw1_t, 512, (wave * 2 + 1) * 16, k0, lane), acc1);
      }
      int n0 = wave * 32 + n15;
      float b0 = GCF(pb1)[n0], b1 = GCF(pb1)[n0 + 16];
#pragma unroll
      for (int j = 0; j < 8; ++j) {
        sHid1[(mb + j) * 512 + n0]      = (__bf16)eluf(acc0[j] + b0);
        sHid1[(mb + j) * 512 + n0 + 16] = (__bf16)eluf(acc1[j] + b1);
      }
    }
    __syncthreads();
    if (wave < 4) {
      v8f acc = z8();
      for (int k0 = 0; k0 < 512; k0 += 32)
        acc = wmma_bf16(load_a(sHid1, 512, k0, lane), load_b(pwo_t, 512, wave * 16, k0, lane), acc);
      int n = wave * 16 + n15; float b = GCF(pbo)[n];
#pragma unroll
      for (int j = 0; j < 8; ++j) sOut[(mb + j) * 64 + n] = acc[j] + b;
    }
    __syncthreads();
    for (int i = tid; i < 512; i += 512) {
      int m = i >> 5, c = i & 31;
      sPM[i] = sOut[m * 64 + c];
      sPS[i] = softplusf(sOut[m * 64 + 32 + c]) + 1e-4f;
    }
    __syncthreads();

    // ---- posterior MLP: [h,e] (K=1280) -> 512 -> 512 -> 64 ----
    {
      const float* eb = emb + ((size_t)t * CB + row0) * CEMB;
      v8f acc0 = z8(), acc1 = z8();
      for (int k0 = 0; k0 < 1280; k0 += 32) {
        v16bf a = (k0 < 256) ? load_a(sHbf, 256, k0, lane)
                             : load_a_f32(eb, CEMB, k0 - 256, lane);
        acc0 = wmma_bf16(a, load_b(qw0_t, 1280, (wave * 2 + 0) * 16, k0, lane), acc0);
        acc1 = wmma_bf16(a, load_b(qw0_t, 1280, (wave * 2 + 1) * 16, k0, lane), acc1);
      }
      int n0 = wave * 32 + n15;
      float b0 = GCF(qb0)[n0], b1 = GCF(qb0)[n0 + 16];
#pragma unroll
      for (int j = 0; j < 8; ++j) {
        sHid0[(mb + j) * 512 + n0]      = (__bf16)eluf(acc0[j] + b0);
        sHid0[(mb + j) * 512 + n0 + 16] = (__bf16)eluf(acc1[j] + b1);
      }
    }
    __syncthreads();
    {
      v8f acc0 = z8(), acc1 = z8();
      for (int k0 = 0; k0 < 512; k0 += 32) {
        v16bf a = load_a(sHid0, 512, k0, lane);
        acc0 = wmma_bf16(a, load_b(qw1_t, 512, (wave * 2 + 0) * 16, k0, lane), acc0);
        acc1 = wmma_bf16(a, load_b(qw1_t, 512, (wave * 2 + 1) * 16, k0, lane), acc1);
      }
      int n0 = wave * 32 + n15;
      float b0 = GCF(qb1)[n0], b1 = GCF(qb1)[n0 + 16];
#pragma unroll
      for (int j = 0; j < 8; ++j) {
        sHid1[(mb + j) * 512 + n0]      = (__bf16)eluf(acc0[j] + b0);
        sHid1[(mb + j) * 512 + n0 + 16] = (__bf16)eluf(acc1[j] + b1);
      }
    }
    __syncthreads();
    if (wave < 4) {
      v8f acc = z8();
      for (int k0 = 0; k0 < 512; k0 += 32)
        acc = wmma_bf16(load_a(sHid1, 512, k0, lane), load_b(qwo_t, 512, wave * 16, k0, lane), acc);
      int n = wave * 16 + n15; float b = GCF(qbo)[n];
#pragma unroll
      for (int j = 0; j < 8; ++j) sOut[(mb + j) * 64 + n] = acc[j] + b;
    }
    __syncthreads();

    // ---- sample s, KL ----
    for (int i = tid; i < 512; i += 512) {
      int m = i >> 5, c = i & 31;
      float qm = sOut[m * 64 + c];
      float qs = softplusf(sOut[m * 64 + 32 + c]) + 1e-4f;
      float nz = GCF(noise)[((size_t)t * CB + row0 + m) * CSTOCH + c];
      float s  = qm + qs * nz;
      sXA[m * 64 + c] = (__bf16)s;
      float pm = sPM[i], ps = sPS[i];
      float dm = qm - pm;
      klacc += logf(ps / qs) + (qs * qs + dm * dm) / (2.f * ps * ps) - 0.5f;
    }
    __syncthreads();

    // ---- write states [h(256), s(32)] as bf16 ----
    for (int idx = tid; idx < 16 * 288; idx += 512) {
      int m = idx / 288, c = idx - m * 288;
      __bf16 v = (c < 256) ? sHbf[m * 256 + c] : sXA[m * 64 + (c - 256)];
      GB(states)[((size_t)t * CB + row0 + m) * 288 + c] = v;
    }
    __syncthreads();
  }

  // deterministic per-block KL partial
#pragma unroll
  for (int off = 16; off > 0; off >>= 1) klacc += __shfl_down(klacc, off, 32);
  if (lane == 0) sRed[wave] = klacc;
  __syncthreads();
  if (tid == 0) {
    float s = 0.f;
    for (int w = 0; w < 16; ++w) s += sRed[w];
    GF(kl_part)[blockIdx.x] = s;
  }
}

// -------- decoder: hidden GEMM + bias + ELU (bf16 out), 4 N-tiles per A load --
__global__ __launch_bounds__(256) void k_gemm_hidden(
    const __bf16* __restrict__ A, const __bf16* __restrict__ WT,
    const float* __restrict__ bias, __bf16* __restrict__ C,
    int M, int N, int K) {
  const int wave = threadIdx.x >> 5, lane = threadIdx.x & 31;
  const int n15 = lane & 15, mbse = (lane >> 4) * 8;
  const int nq = N >> 6;                 // quads of 4 n-tiles
  const int quads = (M >> 4) * nq;
  for (int q = blockIdx.x * 8 + wave; q < quads; q += gridDim.x * 8) {
    int mi = q / nq, qi = q - mi * nq;
    const __bf16* At = A + (size_t)mi * 16 * K;
    v8f acc[4];
#pragma unroll
    for (int j2 = 0; j2 < 4; ++j2) acc[j2] = z8();
    for (int k0 = 0; k0 < K; k0 += 32) {
      v16bf a = load_a_g(At, K, k0, lane);
#pragma unroll
      for (int tt = 0; tt < 4; ++tt)
        acc[tt] = wmma_bf16(a, load_b(WT, K, (qi * 4 + tt) * 16, k0, lane), acc[tt]);
    }
#pragma unroll
    for (int tt = 0; tt < 4; ++tt) {
      int n = (qi * 4 + tt) * 16 + n15;
      float b = GCF(bias)[n];
#pragma unroll
      for (int j = 0; j < 8; ++j)
        GB(C)[((size_t)mi * 16 + mbse + j) * N + n] = (__bf16)eluf(acc[tt][j] + b);
    }
  }
}

// -------- decoder: emb output layer (fp32 out + NLL partial), quad tiles ------
__global__ __launch_bounds__(256) void k_emb_final(
    const __bf16* __restrict__ A, const __bf16* __restrict__ WT,
    const float* __restrict__ bias, const float* __restrict__ embflat,
    float* __restrict__ pred, float* __restrict__ part) {
  __shared__ float sred[8];
  const int wave = threadIdx.x >> 5, lane = threadIdx.x & 31;
  const int n15 = lane & 15, mbse = (lane >> 4) * 8;
  const int nq = CEMB >> 6, quads = ((CT * CB) >> 4) * nq;
  float lp = 0.f;
  for (int q = blockIdx.x * 8 + wave; q < quads; q += gridDim.x * 8) {
    int mi = q / nq, qi = q - mi * nq;
    const __bf16* At = A + (size_t)mi * 16 * CFEAT;
    v8f acc[4];
#pragma unroll
    for (int j2 = 0; j2 < 4; ++j2) acc[j2] = z8();
    for (int k0 = 0; k0 < CFEAT; k0 += 32) {
      v16bf a = load_a_g(At, CFEAT, k0, lane);
#pragma unroll
      for (int tt = 0; tt < 4; ++tt)
        acc[tt] = wmma_bf16(a, load_b(WT, CFEAT, (qi * 4 + tt) * 16, k0, lane), acc[tt]);
    }
#pragma unroll
    for (int tt = 0; tt < 4; ++tt) {
      int n = (qi * 4 + tt) * 16 + n15;
      float b = GCF(bias)[n];
#pragma unroll
      for (int j = 0; j < 8; ++j) {
        size_t r = (size_t)mi * 16 + mbse + j;
        float mean = acc[tt][j] + b;
        GF(pred)[r * CEMB + n] = mean;
        float d = GCF(embflat)[r * CEMB + n] - mean;
        lp += -0.5f * d * d - 0.5f * LOG2PI;
      }
    }
  }
#pragma unroll
  for (int off = 16; off > 0; off >>= 1) lp += __shfl_down(lp, off, 32);
  if (lane == 0) sred[wave] = lp;
  __syncthreads();
  if (threadIdx.x == 0) {
    float s = 0.f;
    for (int w = 0; w < 8; ++w) s += sred[w];
    GF(part)[blockIdx.x] = s;
  }
}

// ---------------- decoder: reward output (N=1 dot product) --------------------
__global__ __launch_bounds__(256) void k_rew_final(
    const __bf16* __restrict__ H, const float* __restrict__ wo,
    const float* __restrict__ bo, const float* __restrict__ reward,
    float* __restrict__ pred, float* __restrict__ part) {
  __shared__ float sred[8];
  int r = blockIdx.x * 256 + threadIdx.x;
  float acc = 0.f;
  for (int k = 0; k < CFEAT; ++k) acc += (float)GCB(H)[(size_t)r * CFEAT + k] * GCF(wo)[k];
  float mean = acc + GCF(bo)[0];
  GF(pred)[r] = mean;
  float d = GCF(reward)[r] - mean;
  float lp = -0.5f * d * d - 0.5f * LOG2PI;
#pragma unroll
  for (int off = 16; off > 0; off >>= 1) lp += __shfl_down(lp, off, 32);
  int lane = threadIdx.x & 31, wave = threadIdx.x >> 5;
  if (lane == 0) sred[wave] = lp;
  __syncthreads();
  if (threadIdx.x == 0) {
    float s = 0.f;
    for (int w = 0; w < 8; ++w) s += sred[w];
    GF(part)[blockIdx.x] = s;
  }
}

__global__ void k_finalize(const float* kl, const float* ep, const float* rp, float* out) {
  if (threadIdx.x == 0 && blockIdx.x == 0) {
    float k = 0.f, e = 0.f, r = 0.f;
    for (int i = 0; i < 8; ++i)   k += GCF(kl)[i];
    for (int i = 0; i < 512; ++i) e += GCF(ep)[i];
    for (int i = 0; i < 64; ++i)  r += GCF(rp)[i];
    GF(out)[0] = (k - e - r) / (float)(CT * CB);
  }
}

extern "C" void kernel_launch(void* const* d_in, const int* in_sizes, int n_in,
                              void* d_out, int out_size, void* d_ws, size_t ws_size,
                              hipStream_t stream) {
  const float* emb    = (const float*)d_in[0];
  const float* action = (const float*)d_in[1];
  const float* reward = (const float*)d_in[2];
  const float* noise  = (const float*)d_in[3];
  const float* g_wih  = (const float*)d_in[4];
  const float* g_bih  = (const float*)d_in[5];
  const float* g_whh  = (const float*)d_in[6];
  const float* g_bhh  = (const float*)d_in[7];
  const float* po_w0  = (const float*)d_in[8];
  const float* po_b0  = (const float*)d_in[9];
  const float* po_w1  = (const float*)d_in[10];
  const float* po_b1  = (const float*)d_in[11];
  const float* po_wo  = (const float*)d_in[12];
  const float* po_bo  = (const float*)d_in[13];
  const float* pr_w0  = (const float*)d_in[14];
  const float* pr_b0  = (const float*)d_in[15];
  const float* pr_w1  = (const float*)d_in[16];
  const float* pr_b1  = (const float*)d_in[17];
  const float* pr_wo  = (const float*)d_in[18];
  const float* pr_bo  = (const float*)d_in[19];
  const float* e_w0   = (const float*)d_in[20];
  const float* e_b0   = (const float*)d_in[21];
  const float* e_w1   = (const float*)d_in[22];
  const float* e_b1   = (const float*)d_in[23];
  const float* e_wo   = (const float*)d_in[24];
  const float* e_bo   = (const float*)d_in[25];
  const float* r_w0   = (const float*)d_in[26];
  const float* r_b0   = (const float*)d_in[27];
  const float* r_w1   = (const float*)d_in[28];
  const float* r_b1   = (const float*)d_in[29];
  const float* r_wo   = (const float*)d_in[30];
  const float* r_bo   = (const float*)d_in[31];

  char* ws = (char*)d_ws;
  size_t off = 0;
  auto carve = [&](size_t bytes) -> void* {
    void* p = ws + off;
    off = (off + bytes + 255) & ~(size_t)255;
    return p;
  };
  __bf16* wihT = (__bf16*)carve(768 * 64 * 2);
  __bf16* whhT = (__bf16*)carve(768 * 256 * 2);
  __bf16* qw0T = (__bf16*)carve((size_t)512 * 1280 * 2);
  __bf16* qw1T = (__bf16*)carve(512 * 512 * 2);
  __bf16* qwoT = (__bf16*)carve(64 * 512 * 2);
  __bf16* pw0T = (__bf16*)carve(512 * 256 * 2);
  __bf16* pw1T = (__bf16*)carve(512 * 512 * 2);
  __bf16* pwoT = (__bf16*)carve(64 * 512 * 2);
  __bf16* ew0T = (__bf16*)carve(512 * 288 * 2);
  __bf16* ew1T = (__bf16*)carve(512 * 512 * 2);
  __bf16* ewoT = (__bf16*)carve((size_t)1024 * 512 * 2);
  __bf16* rw0T = (__bf16*)carve(512 * 288 * 2);
  __bf16* rw1T = (__bf16*)carve(512 * 512 * 2);
  __bf16* states = (__bf16*)carve((size_t)16384 * 288 * 2);
  __bf16* H1 = (__bf16*)carve((size_t)16384 * 512 * 2);
  __bf16* H2 = (__bf16*)carve((size_t)16384 * 512 * 2);
  float* kl_part  = (float*)carve(8 * 4);
  float* emb_part = (float*)carve(512 * 4);
  float* rew_part = (float*)carve(64 * 4);

  auto tl = [&](const float* W, __bf16* WT, int K, int N, int Kpad) {
    int total = N * Kpad;
    k_transpose<<<(total + 255) / 256, 256, 0, stream>>>(W, WT, K, N, Kpad);
  };
  tl(g_wih, wihT, 44, 768, 64);
  tl(g_whh, whhT, 256, 768, 256);
  tl(po_w0, qw0T, 1280, 512, 1280);
  tl(po_w1, qw1T, 512, 512, 512);
  tl(po_wo, qwoT, 512, 64, 512);
  tl(pr_w0, pw0T, 256, 512, 256);
  tl(pr_w1, pw1T, 512, 512, 512);
  tl(pr_wo, pwoT, 512, 64, 512);
  tl(e_w0, ew0T, 288, 512, 288);
  tl(e_w1, ew1T, 512, 512, 512);
  tl(e_wo, ewoT, 512, 1024, 512);
  tl(r_w0, rw0T, 288, 512, 288);
  tl(r_w1, rw1T, 512, 512, 512);

  // sequential scan: 8 blocks (16 batch rows each), 16 waves, full T loop inside
  k_scan<<<8, 512, 0, stream>>>(emb, action, noise,
                                wihT, g_bih, whhT, g_bhh,
                                qw0T, po_b0, qw1T, po_b1, qwoT, po_bo,
                                pw0T, pr_b0, pw1T, pr_b1, pwoT, pr_bo,
                                states, kl_part);

  float* pred_emb = (float*)d_out + 1;
  float* pred_rew = (float*)d_out + 1 + (size_t)CT * CB * CEMB;

  // emb decoder branch
  k_gemm_hidden<<<512, 256, 0, stream>>>(states, ew0T, e_b0, H1, 16384, 512, 288);
  k_gemm_hidden<<<512, 256, 0, stream>>>(H1, ew1T, e_b1, H2, 16384, 512, 512);
  k_emb_final<<<512, 256, 0, stream>>>(H2, ewoT, e_bo, emb, pred_emb, emb_part);

  // reward decoder branch (reuses H1/H2)
  k_gemm_hidden<<<512, 256, 0, stream>>>(states, rw0T, r_b0, H1, 16384, 512, 288);
  k_gemm_hidden<<<512, 256, 0, stream>>>(H1, rw1T, r_b1, H2, 16384, 512, 512);
  k_rew_final<<<64, 256, 0, stream>>>(H2, r_wo, r_bo, reward, pred_rew, rew_part);

  k_finalize<<<1, 32, 0, stream>>>(kl_part, emb_part, rew_part, (float*)d_out);
}
